// GNNEncoder_88587995447837
// MI455X (gfx1250) — matrix operand order repaired and checked by reading, not compiled, not run on hardware
//
#include <hip/hip_runtime.h>

typedef __attribute__((ext_vector_type(16))) __bf16 v16bf;
typedef __attribute__((ext_vector_type(8)))  float  v8f;

#define D_IN  128
#define D_H   256

static __device__ __forceinline__ void split_bf16(float x, __bf16& hi, __bf16& lo) {
    hi = (__bf16)x;
    lo = (__bf16)(x - (float)hi);
}

// ---------------- weight pre-split: fp32 -> (hi, lo) bf16 ----------------
__global__ __launch_bounds__(256) void cvt_w_kernel(const float* __restrict__ W,
                                                    __bf16* __restrict__ Whi,
                                                    __bf16* __restrict__ Wlo, int n) {
    int i = blockIdx.x * 256 + threadIdx.x;
    if (i < n) {
        __bf16 h, l;
        split_bf16(W[i], h, l);
        Whi[i] = h;
        Wlo[i] = l;
    }
}

// ---------------- degree / normalization ----------------
__global__ __launch_bounds__(256) void deg_init_kernel(int* __restrict__ degi, int n) {
    int i = blockIdx.x * 256 + threadIdx.x;
    if (i < n) degi[i] = 1;  // self loop
}

__global__ __launch_bounds__(256) void deg_edge_kernel(const int* __restrict__ dst,
                                                       int* __restrict__ degi, int e) {
    int i = blockIdx.x * 256 + threadIdx.x;
    if (i < e) atomicAdd(&degi[dst[i]], 1);
}

__global__ __launch_bounds__(256) void deg_inv_kernel(const int* __restrict__ degi,
                                                      float* __restrict__ dinv, int n) {
    int i = blockIdx.x * 256 + threadIdx.x;
    if (i < n) dinv[i] = rsqrtf((float)degi[i]);
}

// ---------------- GEMM: H[M x 256] = X[M x K] @ W[K x 256] ----------------
// fp32 emulated via 3x bf16 WMMA: X*W ~= Xhi*Whi + Xhi*Wlo + Xlo*Whi
// One wave -> one 16x16 tile. grid = (M/16, 256/16/8), block = 256 (8 waves).
__global__ __launch_bounds__(256) void gemm_b3_kernel(const float* __restrict__ X,
                                                      const __bf16* __restrict__ Whi,
                                                      const __bf16* __restrict__ Wlo,
                                                      float* __restrict__ H,
                                                      int M, int K) {
    const int lane    = threadIdx.x & 31;
    const int wave    = threadIdx.x >> 5;                 // 0..7
    const int mtile   = blockIdx.x;                       // 16 rows
    const int ntile   = blockIdx.y * 8 + wave;            // 16 cols
    const int row     = mtile * 16 + (lane & 15);
    const int n0      = ntile * 16;
    const bool hihalf = lane >= 16;
    const int  koff   = hihalf ? 8 : 0;

    v8f acc = {};
    for (int k0 = 0; k0 < K; k0 += 32) {
        // ---- A fragment (16x32 bf16, ISA layout): lane holds row = lane&15,
        //      halves 0..7 -> k0+koff+0..7, halves 8..15 -> k0+16+koff+0..7
        const float* xp = X + (size_t)row * K + k0 + koff;
        const float4 a0 = *(const float4*)(xp + 0);
        const float4 a1 = *(const float4*)(xp + 4);
        const float4 a2 = *(const float4*)(xp + 16);
        const float4 a3 = *(const float4*)(xp + 20);
        float xa[16] = {a0.x, a0.y, a0.z, a0.w, a1.x, a1.y, a1.z, a1.w,
                        a2.x, a2.y, a2.z, a2.w, a3.x, a3.y, a3.z, a3.w};
        v16bf ahi, alo;
#pragma unroll
        for (int i = 0; i < 16; ++i) {
            __bf16 h, l;
            split_bf16(xa[i], h, l);
            ahi[i] = h;
            alo[i] = l;
        }
        // ---- B fragment (32x16 bf16): lane holds col n0+(lane&15), halves follow A's k order
        const __bf16* wph = Whi + (size_t)(k0 + koff) * D_H + n0 + (lane & 15);
        const __bf16* wpl = Wlo + (size_t)(k0 + koff) * D_H + n0 + (lane & 15);
        v16bf bhi, blo;
#pragma unroll
        for (int i = 0; i < 16; ++i) {
            const int kk = (i < 8) ? i : i + 8;
            bhi[i] = wph[(size_t)kk * D_H];
            blo[i] = wpl[(size_t)kk * D_H];
        }

        acc = __builtin_amdgcn_wmma_f32_16x16x32_bf16(false, ahi, false, bhi, (short)0, acc, false, false);
        acc = __builtin_amdgcn_wmma_f32_16x16x32_bf16(false, ahi, false, blo, (short)0, acc, false, false);
        acc = __builtin_amdgcn_wmma_f32_16x16x32_bf16(false, alo, false, bhi, (short)0, acc, false, false);
    }

    // ---- C/D layout: VGPR i -> row mtile*16 + i (+8 for high half-wave), col = n0 + lane&15
    const int crow = mtile * 16 + (hihalf ? 8 : 0);
    const int ccol = n0 + (lane & 15);
#pragma unroll
    for (int i = 0; i < 8; ++i) {
        H[(size_t)(crow + i) * D_H + ccol] = acc[i];
    }
}

// ---------------- aggregation ----------------
// ACC[i][:] = H[i][:] * dinv[i]^2   (self-loop contribution, also inits buffer)
__global__ __launch_bounds__(256) void self_init_kernel(const float* __restrict__ H,
                                                        const float* __restrict__ dinv,
                                                        float* __restrict__ ACC, int total4) {
    int t = blockIdx.x * 256 + threadIdx.x;  // one float4 per thread
    if (t < total4) {
        int node = t >> 6;  // 64 float4 per node (256 feats)
        float w = dinv[node] * dinv[node];
        float4 v = ((const float4*)H)[t];
        v.x *= w; v.y *= w; v.z *= w; v.w *= w;
        ((float4*)ACC)[t] = v;
    }
}

// Native no-return f32 atomic RMW at device scope (resolves in L2).
// Guarantees GLOBAL_ATOMIC_ADD_F32 instead of a possible CAS retry loop.
static __device__ __forceinline__ void atomAddF32(float* p, float v) {
    asm volatile("global_atomic_add_f32 %0, %1, off scope:SCOPE_DEV"
                 :
                 : "v"(p), "v"(v)
                 : "memory");
}

// One 64-thread group per edge; each thread owns 4 features (float4 gather, 4 f32 atomics).
__global__ __launch_bounds__(256) void edge_agg_kernel(const int* __restrict__ src,
                                                       const int* __restrict__ dst,
                                                       const float* __restrict__ dinv,
                                                       const float* __restrict__ H,
                                                       float* __restrict__ ACC, int E) {
    int t = blockIdx.x * 256 + threadIdx.x;
    int e = t >> 6;
    if (e >= E) return;
    int f = (t & 63) << 2;
    int s = src[e], d = dst[e];
    float w = dinv[s] * dinv[d];
    float4 v = *(const float4*)(H + (size_t)s * D_H + f);
    float* out = ACC + (size_t)d * D_H + f;
    atomAddF32(out + 0, v.x * w);
    atomAddF32(out + 1, v.y * w);
    atomAddF32(out + 2, v.z * w);
    atomAddF32(out + 3, v.w * w);
}

// OUT = relu(ACC + b)
__global__ __launch_bounds__(256) void finalize_kernel(const float* __restrict__ ACC,
                                                       const float* __restrict__ b,
                                                       float* __restrict__ OUT, int total4) {
    int t = blockIdx.x * 256 + threadIdx.x;
    if (t < total4) {
        int f = (t & 63) << 2;
        float4 v = ((const float4*)ACC)[t];
        float4 bb = *(const float4*)(b + f);
        v.x = fmaxf(v.x + bb.x, 0.0f);
        v.y = fmaxf(v.y + bb.y, 0.0f);
        v.z = fmaxf(v.z + bb.z, 0.0f);
        v.w = fmaxf(v.w + bb.w, 0.0f);
        ((float4*)OUT)[t] = v;
    }
}

extern "C" void kernel_launch(void* const* d_in, const int* in_sizes, int n_in,
                              void* d_out, int out_size, void* d_ws, size_t ws_size,
                              hipStream_t stream) {
    const float* X  = (const float*)d_in[0];
    const int*   EI = (const int*)d_in[1];
    const float* W1 = (const float*)d_in[2];
    const float* B1 = (const float*)d_in[3];
    const float* W2 = (const float*)d_in[4];
    const float* B2 = (const float*)d_in[5];
    float* OUT = (float*)d_out;

    const int N = in_sizes[0] / D_IN;  // 100000
    const int E = in_sizes[1] / 2;     // 1600000
    const int* src = EI;
    const int* dst = EI + E;

    // ---- workspace carving (256B aligned regions) ----
    char* ws = (char*)d_ws;
    size_t off = 0;
    auto alloc = [&](size_t bytes) -> char* {
        char* p = ws + off;
        off = (off + bytes + 255) & ~(size_t)255;
        return p;
    };
    int*    degi = (int*)alloc((size_t)N * 4);
    float*  dinv = (float*)alloc((size_t)N * 4);
    __bf16* W1hi = (__bf16*)alloc((size_t)D_IN * D_H * 2);
    __bf16* W1lo = (__bf16*)alloc((size_t)D_IN * D_H * 2);
    __bf16* W2hi = (__bf16*)alloc((size_t)D_H * D_H * 2);
    __bf16* W2lo = (__bf16*)alloc((size_t)D_H * D_H * 2);
    float*  bufA = (float*)alloc((size_t)N * D_H * 4);   // 102.4 MB
    float*  bufB = (float*)alloc((size_t)N * D_H * 4);   // 102.4 MB

    const int nblk   = (N + 255) / 256;
    const int eblk   = (E + 255) / 256;
    const int total4 = N * (D_H / 4);
    const int fblk   = (total4 + 255) / 256;
    const int eat    = (E * (D_H / 4) + 255) / 256;      // edge-agg thread blocks
    const dim3 ggrid((N + 15) / 16, D_H / 16 / 8);       // (6250, 2)

    // weight split (tiny)
    cvt_w_kernel<<<(D_IN * D_H + 255) / 256, 256, 0, stream>>>(W1, W1hi, W1lo, D_IN * D_H);
    cvt_w_kernel<<<(D_H * D_H + 255) / 256, 256, 0, stream>>>(W2, W2hi, W2lo, D_H * D_H);

    // symmetric normalization coefficients
    deg_init_kernel<<<nblk, 256, 0, stream>>>(degi, N);
    deg_edge_kernel<<<eblk, 256, 0, stream>>>(dst, degi, E);
    deg_inv_kernel<<<nblk, 256, 0, stream>>>(degi, dinv, N);

    // ---- layer 1 ----
    gemm_b3_kernel<<<ggrid, 256, 0, stream>>>(X, W1hi, W1lo, bufA, N, D_IN);      // bufA = X @ W1
    self_init_kernel<<<fblk, 256, 0, stream>>>(bufA, dinv, bufB, total4);         // bufB = self-loop
    edge_agg_kernel<<<eat, 256, 0, stream>>>(src, dst, dinv, bufA, bufB, E);      // scatter-add
    finalize_kernel<<<fblk, 256, 0, stream>>>(bufB, B1, bufA, total4);            // bufA = relu(.+b1)

    // ---- layer 2 ----
    gemm_b3_kernel<<<ggrid, 256, 0, stream>>>(bufA, W2hi, W2lo, bufB, N, D_H);    // bufB = A1 @ W2
    self_init_kernel<<<fblk, 256, 0, stream>>>(bufB, dinv, bufA, total4);
    edge_agg_kernel<<<eat, 256, 0, stream>>>(src, dst, dinv, bufB, bufA, E);
    finalize_kernel<<<fblk, 256, 0, stream>>>(bufA, B2, OUT, total4);
}
